// DecoderRNN_11115375362540
// MI455X (gfx1250) — compile-verified, matrix operand order checked by reading, not run on hardware
//
#include <hip/hip_runtime.h>
#include <hip/hip_bf16.h>

// ---------------------------------------------------------------------------
// DecoderRNN (Bahdanau attention + LSTMCell + vocab projection) for gfx1250.
// All GEMMs run through v_wmma_f32_16x16x32_bf16 with f32 accumulation.
// GEMM uses double-buffered LDS tiles; staging prefers the CDNA5 async
// global->LDS DMA path (ASYNCcnt) and falls back to a register pipeline.
// ---------------------------------------------------------------------------

typedef __attribute__((ext_vector_type(16))) __bf16 v16bf;
typedef __attribute__((ext_vector_type(8)))  float  v8f;

#if defined(__has_builtin)
#if __has_builtin(__builtin_amdgcn_global_load_async_to_lds_b128)
#define USE_ASYNC_LDS 1
#endif
#endif
#ifndef USE_ASYNC_LDS
#define USE_ASYNC_LDS 0
#endif

#if USE_ASYNC_LDS
// Builtin signature (from compiler diagnostic): param0 = int4 in AS1 (global),
// param1 = int4 in AS3 (LDS), then two integer immediates (offset, cpol).
typedef int v4i_t __attribute__((vector_size(4 * sizeof(int))));
typedef __attribute__((address_space(1))) v4i_t as1_v4i;
typedef __attribute__((address_space(3))) v4i_t as3_v4i;
#endif

__device__ __forceinline__ void wait_async0() {
#if USE_ASYNC_LDS
#if __has_builtin(__builtin_amdgcn_s_wait_asynccnt)
    __builtin_amdgcn_s_wait_asynccnt(0);
#else
    asm volatile("s_wait_asynccnt 0x0" ::: "memory");
#endif
#endif
}

#define B_  128
#define T_  25
#define L_  49
#define V_  10000
#define E_  512
#define H_  512
#define ENC_ 512

// ---------------------------------------------------------------------------
// Fragment load from an LDS tile (row-major, ld = 32 bf16 elements).
// CDNA5 ISA 7.12.2 layout for 16-bit A/B operands (16 rows x 32 K):
//   lanes 0-15 : row = base+lane,    K = {0..7} and {16..23}
//   lanes 16-31: row = base+lane-16, K = {8..15} and {24..31}
// Emits two ds_load_b128 per fragment.
// ---------------------------------------------------------------------------
__device__ __forceinline__ v16bf lds_frag(const __bf16* __restrict__ base,
                                          int row, int lane) {
    const __bf16* q = base + (row + (lane & 15)) * 32 + ((lane >> 4) << 3);
    v16bf f;
#pragma unroll
    for (int j = 0; j < 8; ++j) f[j] = q[j];          // K = kk+0..7
#pragma unroll
    for (int j = 0; j < 8; ++j) f[8 + j] = q[16 + j]; // K = kk+16..23
    return f;
}

// One K-block of compute: 1 A-fragment reused over 4 B-fragments / WMMAs.
__device__ __forceinline__ void wmma_block(const __bf16* sAbuf, const __bf16* sBbuf,
                                           int wm, int wn, int lane, v8f acc[4]) {
    v16bf af = lds_frag(sAbuf, wm * 16, lane);
#pragma unroll
    for (int nt = 0; nt < 4; ++nt) {
        v16bf bf = lds_frag(sBbuf, wn * 64 + nt * 16, lane);
        acc[nt] = __builtin_amdgcn_wmma_f32_16x16x32_bf16(
            false, af, false, bf, (short)0, acc[nt], false, false);
    }
}

// C/D tile store per ISA: VGPR i, lane l -> row = i + 8*(l>>4), col = l&15.
__device__ __forceinline__ void store_tile(float* __restrict__ C, size_t ldc,
                                           int row, int col, int M, int N,
                                           v8f acc, const float* __restrict__ bias,
                                           int accumulate, int lane) {
    int hi = lane >> 4;
    int c  = col + (lane & 15);
    if (c >= N) return;
    float bv = bias ? bias[c] : 0.0f;
#pragma unroll
    for (int i = 0; i < 8; ++i) {
        int r = row + i + 8 * hi;
        if (r < M) {
            size_t off = (size_t)r * ldc + c;
            float v = acc[i] + bv;
            if (accumulate) v += C[off];
            C[off] = v;
        }
    }
}

// ---------------------------------------------------------------------------
// GEMM:  C[M,N] (f32, ld=ldc) (+)= A[M,K] * B[N,K]^T   (A,B bf16 row-major)
// Block: 256 threads (8 waves), tile 64(M) x 128(N), K-step 32.
// Waves: 4 along M x 2 along N; each wave owns 16x64 (4 accumulators).
// LDS double buffer: sA 2x64x32 (8KB), sB 2x128x32 (16KB). One barrier per
// K-iteration; next tile streams in (async DMA or reg pipeline) during WMMA.
// ---------------------------------------------------------------------------
__global__ __launch_bounds__(256)
void gemm_bt_wmma(const __bf16* __restrict__ A, const __bf16* __restrict__ Bm,
                  float* __restrict__ C, size_t ldc,
                  int M, int N, int K,
                  const float* __restrict__ bias, int accumulate) {
    __shared__ alignas(16) __bf16 sA[2][64 * 32];
    __shared__ alignas(16) __bf16 sB[2][128 * 32];

    const int tid  = threadIdx.x;
    const int lane = tid & 31;
    const int wave = tid >> 5;
    const int wm   = wave & 3;   // 16-row stripe within 64
    const int wn   = wave >> 2;  // 64-col stripe within 128
    const int bm   = blockIdx.y * 64;
    const int bn   = blockIdx.x * 128;

    // Cooperative global->LDS mapping (all 16B-aligned chunks):
    //   A tile: 64 rows x 64B  -> 4 threads/row, 1 x b128 each
    //   B tile: 128 rows x 64B -> 2 threads/row, 2 x b128 each
    const int arow = tid >> 2;
    const int acol = (tid & 3) * 8;        // bf16 elements
    const int brow = tid >> 1;
    const int bcol = (tid & 1) * 16;       // bf16 elements

    int agr = bm + arow; if (agr > M - 1) agr = M - 1;   // clamped rows are
    int bgr = bn + brow; if (bgr > N - 1) bgr = N - 1;   // never stored
    const __bf16* aptr = A  + (size_t)agr * K + acol;
    const __bf16* bptr = Bm + (size_t)bgr * K + bcol;

    v8f acc[4] = {};
    const int nk = K >> 5;   // K is always a multiple of 32 here

#if USE_ASYNC_LDS
    // ---- async DMA staging (GLOBAL_LOAD_ASYNC_TO_LDS_B128, ASYNCcnt) ----
    auto stage = [&](int buf, int k0) {
        __builtin_amdgcn_global_load_async_to_lds_b128(
            (as1_v4i*)(aptr + k0),
            (as3_v4i*)&sA[buf][arow * 32 + acol], 0, 0);
        __builtin_amdgcn_global_load_async_to_lds_b128(
            (as1_v4i*)(bptr + k0),
            (as3_v4i*)&sB[buf][brow * 32 + bcol], 0, 0);
        __builtin_amdgcn_global_load_async_to_lds_b128(
            (as1_v4i*)(bptr + k0 + 8),
            (as3_v4i*)&sB[buf][brow * 32 + bcol + 8], 0, 0);
    };
    stage(0, 0);
    wait_async0();
    __syncthreads();
    for (int kt = 0; kt < nk; ++kt) {
        const int  cur      = kt & 1;
        const bool has_next = (kt + 1 < nk);
        if (has_next) stage(cur ^ 1, (kt + 1) << 5);   // DMA overlaps WMMAs
        wmma_block(sA[cur], sB[cur], wm, wn, lane, acc);
        if (has_next) {
            wait_async0();       // next tile fully in LDS (this wave's DMAs)
            __syncthreads();     // ...and everyone else's
        }
    }
#else
    // ---- register-staged software pipeline (fallback) ----
    uint4 ra  = *(const uint4*)(aptr);
    uint4 rb0 = *(const uint4*)(bptr);
    uint4 rb1 = *(const uint4*)(bptr + 8);
    *(uint4*)&sA[0][arow * 32 + acol]     = ra;
    *(uint4*)&sB[0][brow * 32 + bcol]     = rb0;
    *(uint4*)&sB[0][brow * 32 + bcol + 8] = rb1;
    __syncthreads();
    for (int kt = 0; kt < nk; ++kt) {
        const int  cur      = kt & 1;
        const bool has_next = (kt + 1 < nk);
        uint4 na = make_uint4(0, 0, 0, 0), nb0 = na, nb1 = na;
        if (has_next) {                       // issue next tile's loads NOW
            int k0n = (kt + 1) << 5;
            na  = *(const uint4*)(aptr + k0n);
            nb0 = *(const uint4*)(bptr + k0n);
            nb1 = *(const uint4*)(bptr + k0n + 8);
        }
        wmma_block(sA[cur], sB[cur], wm, wn, lane, acc);
        if (has_next) {                       // commit into the other buffer
            *(uint4*)&sA[cur ^ 1][arow * 32 + acol]     = na;
            *(uint4*)&sB[cur ^ 1][brow * 32 + bcol]     = nb0;
            *(uint4*)&sB[cur ^ 1][brow * 32 + bcol + 8] = nb1;
            __syncthreads();                  // single barrier per iteration
        }
    }
#endif

#pragma unroll
    for (int nt = 0; nt < 4; ++nt)
        store_tile(C, ldc, bm + wm * 16, bn + wn * 64 + nt * 16,
                   M, N, acc[nt], bias, accumulate, lane);
}

// ---------------------------------------------------------------------------
// Prep kernels
// ---------------------------------------------------------------------------
__global__ void f32_to_bf16_kernel(const float* __restrict__ src,
                                   __bf16* __restrict__ dst, int n) {
    for (int i = blockIdx.x * blockDim.x + threadIdx.x; i < n;
         i += gridDim.x * blockDim.x)
        dst[i] = (__bf16)src[i];
}

__global__ void embed_gather_kernel(const int* __restrict__ captions,
                                    const float* __restrict__ table,
                                    __bf16* __restrict__ out) {
    // out[b,t,e] = table[captions[b,t], e]   (padding row 0 is already zero)
    int n = B_ * T_ * E_;
    for (int i = blockIdx.x * blockDim.x + threadIdx.x; i < n;
         i += gridDim.x * blockDim.x) {
        int e  = i & (E_ - 1);
        int bt = i >> 9;
        int tok = captions[bt];
        out[i] = (__bf16)table[(size_t)tok * E_ + e];
    }
}

__global__ void init_state_kernel(__bf16* __restrict__ h_bf,
                                  float* __restrict__ c_state) {
    int i = blockIdx.x * blockDim.x + threadIdx.x;  // B*H = 65536
    if (i < B_ * H_) { h_bf[i] = (__bf16)0.0f; c_state[i] = 0.0f; }
}

// ---------------------------------------------------------------------------
// Attention: one block per batch row.
//   scores_l = sum_h tanh(hidden_proj[b,h] + enc_proj[b,l,h]) * v[h]
//   attn     = softmax(scores);  context = attn @ encF[b]
//   x_cat[b] = [ embedded[b,t,:], context ]  (bf16, feeds the gates GEMM)
// ---------------------------------------------------------------------------
__global__ __launch_bounds__(256)
void attention_step_kernel(const float* __restrict__ hidden_proj, // [B,H]
                           const float* __restrict__ enc_proj,    // [B,L,H]
                           const float* __restrict__ v_attn,      // [H]
                           const float* __restrict__ encF,        // [B,L,ENC]
                           const __bf16* __restrict__ emb,        // [B,T,E]
                           int t,
                           __bf16* __restrict__ x_cat) {          // [B,E+ENC]
    int b = blockIdx.x;
    int tid = threadIdx.x;
    int lane = tid & 31, wave = tid >> 5;

    __shared__ float s_hp[H_];
    __shared__ float s_sc[64];

    for (int h = tid; h < H_; h += 256) s_hp[h] = hidden_proj[b * H_ + h];
    __syncthreads();

    for (int l = wave; l < L_; l += 8) {
        const float* ep = enc_proj + ((size_t)b * L_ + l) * H_;
        float partial = 0.0f;
        for (int h = lane; h < H_; h += 32)
            partial += tanhf(s_hp[h] + ep[h]) * v_attn[h];
#pragma unroll
        for (int off = 16; off > 0; off >>= 1)
            partial += __shfl_down(partial, off);
        if (lane == 0) s_sc[l] = partial;
    }
    __syncthreads();

    if (tid == 0) {  // tiny softmax over L=49
        float m = s_sc[0];
        for (int l = 1; l < L_; ++l) m = fmaxf(m, s_sc[l]);
        float sum = 0.0f;
        for (int l = 0; l < L_; ++l) { float e = __expf(s_sc[l] - m); s_sc[l] = e; sum += e; }
        float inv = 1.0f / sum;
        for (int l = 0; l < L_; ++l) s_sc[l] *= inv;
    }
    __syncthreads();

    const float* ef = encF + (size_t)b * L_ * ENC_;
    for (int h = tid; h < ENC_; h += 256) {
        float ctx = 0.0f;
        for (int l = 0; l < L_; ++l) ctx += s_sc[l] * ef[(size_t)l * ENC_ + h];
        x_cat[(size_t)b * (E_ + ENC_) + E_ + h] = (__bf16)ctx;
        x_cat[(size_t)b * (E_ + ENC_) + h] =
            emb[((size_t)b * T_ + t) * E_ + h];
    }
}

// ---------------------------------------------------------------------------
// LSTM pointwise:  gates layout [i | f | g | o] along 4H (PyTorch order).
// ---------------------------------------------------------------------------
__device__ __forceinline__ float sigmoidf_(float x) {
    return 1.0f / (1.0f + __expf(-x));
}

__global__ void lstm_step_kernel(const float* __restrict__ gates,  // [B,4H]
                                 const float* __restrict__ b_ih,
                                 const float* __restrict__ b_hh,
                                 float* __restrict__ c_state,      // [B,H]
                                 __bf16* __restrict__ h_bf) {      // [B,H]
    int idx = blockIdx.x * blockDim.x + threadIdx.x;  // B*H
    if (idx >= B_ * H_) return;
    int b = idx >> 9, j = idx & (H_ - 1);
    const float* g = gates + (size_t)b * 4 * H_;
    float gi = g[j]            + b_ih[j]            + b_hh[j];
    float gf = g[H_ + j]       + b_ih[H_ + j]       + b_hh[H_ + j];
    float gg = g[2 * H_ + j]   + b_ih[2 * H_ + j]   + b_hh[2 * H_ + j];
    float go = g[3 * H_ + j]   + b_ih[3 * H_ + j]   + b_hh[3 * H_ + j];
    float cn = sigmoidf_(gf) * c_state[idx] + sigmoidf_(gi) * tanhf(gg);
    c_state[idx] = cn;
    h_bf[idx] = (__bf16)(sigmoidf_(go) * tanhf(cn));
}

// ---------------------------------------------------------------------------
// Host launcher
// ---------------------------------------------------------------------------
extern "C" void kernel_launch(void* const* d_in, const int* in_sizes, int n_in,
                              void* d_out, int out_size, void* d_ws, size_t ws_size,
                              hipStream_t stream) {
    const int*   captions  = (const int*)  d_in[0];
    const float* encF      = (const float*)d_in[1];
    const float* emb_table = (const float*)d_in[2];
    const float* W_h       = (const float*)d_in[3];
    const float* W_e       = (const float*)d_in[4];
    const float* v_attn    = (const float*)d_in[5];
    const float* W_ih      = (const float*)d_in[6];
    const float* W_hh      = (const float*)d_in[7];
    const float* b_ih      = (const float*)d_in[8];
    const float* b_hh      = (const float*)d_in[9];
    const float* W_fc      = (const float*)d_in[10];
    const float* b_fc      = (const float*)d_in[11];
    float* out = (float*)d_out;

    // ---- workspace carve-up (256B aligned slices) ----
    char* p = (char*)d_ws;
    auto carve = [&](size_t bytes) {
        char* r = p; p += (bytes + 255) & ~(size_t)255; return r;
    };
    __bf16* emb_bf  = (__bf16*)carve((size_t)B_ * T_ * E_ * 2);
    __bf16* encF_bf = (__bf16*)carve((size_t)B_ * L_ * ENC_ * 2);
    __bf16* Wh_bf   = (__bf16*)carve((size_t)H_ * H_ * 2);
    __bf16* We_bf   = (__bf16*)carve((size_t)H_ * ENC_ * 2);
    __bf16* Wih_bf  = (__bf16*)carve((size_t)4 * H_ * (E_ + ENC_) * 2);
    __bf16* Whh_bf  = (__bf16*)carve((size_t)4 * H_ * H_ * 2);
    __bf16* Wfc_bf  = (__bf16*)carve((size_t)V_ * H_ * 2);
    float*  enc_proj    = (float*)carve((size_t)B_ * L_ * H_ * 4);
    float*  hidden_proj = (float*)carve((size_t)B_ * H_ * 4);
    float*  gates       = (float*)carve((size_t)B_ * 4 * H_ * 4);
    float*  c_state     = (float*)carve((size_t)B_ * H_ * 4);
    __bf16* h_bf        = (__bf16*)carve((size_t)B_ * H_ * 2);
    __bf16* x_cat       = (__bf16*)carve((size_t)B_ * (E_ + ENC_) * 2);

    auto cvt = [&](const float* src, __bf16* dst, int n) {
        int g = (n + 255) / 256; if (g > 4096) g = 4096;
        f32_to_bf16_kernel<<<g, 256, 0, stream>>>(src, dst, n);
    };
    auto gemm = [&](const __bf16* A, const __bf16* Bm, float* C, size_t ldc,
                    int M, int N, int K, const float* bias, int acc) {
        dim3 grid((N + 127) / 128, (M + 63) / 64);
        gemm_bt_wmma<<<grid, 256, 0, stream>>>(A, Bm, C, ldc, M, N, K, bias, acc);
    };

    // ---- one-time prep ----
    cvt(W_h,  Wh_bf,  H_ * H_);
    cvt(W_e,  We_bf,  H_ * ENC_);
    cvt(W_ih, Wih_bf, 4 * H_ * (E_ + ENC_));
    cvt(W_hh, Whh_bf, 4 * H_ * H_);
    cvt(W_fc, Wfc_bf, V_ * H_);
    cvt(encF, encF_bf, B_ * L_ * ENC_);
    embed_gather_kernel<<<4096, 256, 0, stream>>>(captions, emb_table, emb_bf);
    init_state_kernel<<<(B_ * H_ + 255) / 256, 256, 0, stream>>>(h_bf, c_state);

    // enc_proj[B*L, H] = encF_bf[B*L, ENC] @ W_e^T   (loop-invariant)
    gemm(encF_bf, We_bf, enc_proj, (size_t)H_, B_ * L_, H_, ENC_, nullptr, 0);

    // ---- 25 sequential decoder steps ----
    for (int t = 0; t < T_; ++t) {
        // hidden_proj = h @ W_h^T
        gemm(h_bf, Wh_bf, hidden_proj, (size_t)H_, B_, H_, H_, nullptr, 0);
        // attention + softmax + context + x_cat assembly
        attention_step_kernel<<<B_, 256, 0, stream>>>(
            hidden_proj, enc_proj, v_attn, encF, emb_bf, t, x_cat);
        // gates = x_cat @ W_ih^T  ;  gates += h @ W_hh^T
        gemm(x_cat, Wih_bf, gates, (size_t)(4 * H_), B_, 4 * H_, E_ + ENC_, nullptr, 0);
        gemm(h_bf,  Whh_bf, gates, (size_t)(4 * H_), B_, 4 * H_, H_,       nullptr, 1);
        // LSTM pointwise -> updates c_state, h_bf
        lstm_step_kernel<<<(B_ * H_ + 255) / 256, 256, 0, stream>>>(
            gates, b_ih, b_hh, c_state, h_bf);
        // logits[:, t, :] = h_new @ W_fc^T + b_fc   (ldc = T*V, base offset t*V)
        gemm(h_bf, Wfc_bf, out + (size_t)t * V_, (size_t)T_ * V_,
             B_, V_, H_, b_fc, 0);
    }
}
